// AACN_Layer_54803782697086
// MI455X (gfx1250) — compile-verified, matrix operand order checked by reading, not run on hardware
//
#include <hip/hip_runtime.h>

typedef __attribute__((ext_vector_type(16))) __bf16 v16bf;
typedef __attribute__((ext_vector_type(8)))  __bf16 v8bf;
typedef __attribute__((ext_vector_type(8)))  float  v8f;

#define DEV __device__ __forceinline__

DEV v8f wmma_bf16(v16bf a, v16bf b, v8f c) {
    // (neg_a, A, neg_b, B, c_mod, C, reuse_a, reuse_b)
    return __builtin_amdgcn_wmma_f32_16x16x32_bf16(false, a, false, b, (short)0, c,
                                                   false, false);
}

// ISA 7.12.2 (16-bit A/B): lane's 16 K-values = two contiguous runs of 8
// -> two b128 loads when K is contiguous in memory.
DEV v16bf load_frag(const __bf16* row, int half) {
    v8bf lo = *(const v8bf*)(row + half * 8);
    v8bf hi = *(const v8bf*)(row + 16 + half * 8);
    return __builtin_shufflevector(lo, hi, 0, 1, 2, 3, 4, 5, 6, 7,
                                   8, 9, 10, 11, 12, 13, 14, 15);
}

// ------------------------------ pack kernels --------------------------------
__global__ void zero_bf16_kernel(__bf16* __restrict__ p, int n) {
    int i = blockIdx.x * 256 + threadIdx.x;
    if (i < n) p[i] = (__bf16)0.f;
}

__global__ void cvt_bf16_kernel(const float* __restrict__ s, __bf16* __restrict__ d, int n) {
    int i = blockIdx.x * 256 + threadIdx.x;
    if (i < n) d[i] = (__bf16)s[i];
}

// x[ci][s] (f32) -> xT[s][ci] (bf16), ci contiguous
__global__ void pack_xT_kernel(const float* __restrict__ x, __bf16* __restrict__ xT) {
    int i = blockIdx.x * 256 + threadIdx.x;         // 256*4096
    int ci = i >> 12, s = i & 4095;
    xT[s * 256 + ci] = (__bf16)x[i];
}

// conv_w[(m*256+ci)*27+tap] (f32) -> wp[(tap*192+m)*256+ci] (bf16)
__global__ void pack_wconv_kernel(const float* __restrict__ w, __bf16* __restrict__ wp) {
    int i = blockIdx.x * 256 + threadIdx.x;         // 1327104
    if (i < 192 * 256 * 27) {
        int tap = i % 27;
        int t   = i / 27;
        int ci  = t & 255;
        int m   = t >> 8;
        wp[(tap * 192 + m) * 256 + ci] = (__bf16)w[i];
    }
}

// scatter one kqv output value into padded/reshaped q/k/v buffers
DEV void kqv_scatter(float val, int m, int n,
                     __bf16* __restrict__ qb, __bf16* __restrict__ kb,
                     __bf16* __restrict__ vb)
{
    if (m < 160) {                                     // k: (160,4096)->(8,4096,20)
        int f = m * 4096 + n;
        int h = f / 81920, idx = f % 81920;
        kb[((h * 4096) + idx / 20) * 32 + (idx % 20)] = (__bf16)val;
    } else if (m < 320) {                              // q (pre-scaled by 20^-0.5)
        int f = (m - 160) * 4096 + n;
        int h = f / 81920, idx = f % 81920;
        qb[((h * 4096) + idx / 20) * 32 + (idx % 20)] =
            (__bf16)(val * 0.223606797749979f);
    } else {                                           // v transposed: [h][dvh][key]
        int f = (m - 320) * 4096 + n;
        int h = f >> 15, idx = f & 32767;
        vb[(h * 16 + (idx & 7)) * 4096 + (idx >> 3)] = (__bf16)val;
    }
}

// ------- kqv 1x1x1 conv: [384x256] x [256x4096] + bias, 32x32/wave ----------
__global__ void kqv_gemm_kernel(const __bf16* __restrict__ xT,
                                const __bf16* __restrict__ wkqv,
                                const float* __restrict__ bias,
                                __bf16* __restrict__ qb,
                                __bf16* __restrict__ kb,
                                __bf16* __restrict__ vb)
{
    const int lane = threadIdx.x & 31;
    const int half = lane >> 4;
    const int r    = lane & 15;
    const int m0   = blockIdx.x * 32;
    const int n0   = blockIdx.y * 32;

    const __bf16* a0 = wkqv + (m0 + r) * 256;
    const __bf16* a1 = wkqv + (m0 + 16 + r) * 256;
    const __bf16* b0 = xT   + (n0 + r) * 256;
    const __bf16* b1 = xT   + (n0 + 16 + r) * 256;

    v8f c00 = {}, c01 = {}, c10 = {}, c11 = {};
    #pragma unroll
    for (int k0 = 0; k0 < 256; k0 += 32) {
        v16bf af0 = load_frag(a0 + k0, half), af1 = load_frag(a1 + k0, half);
        v16bf bf0 = load_frag(b0 + k0, half), bf1 = load_frag(b1 + k0, half);
        c00 = wmma_bf16(af0, bf0, c00);
        c01 = wmma_bf16(af0, bf1, c01);
        c10 = wmma_bf16(af1, bf0, c10);
        c11 = wmma_bf16(af1, bf1, c11);
    }

    #pragma unroll
    for (int i = 0; i < 8; ++i) {
        int mi = i + 8 * half;
        kqv_scatter(c00[i] + bias[m0 + mi],      m0 + mi,      n0 + r,      qb, kb, vb);
        kqv_scatter(c01[i] + bias[m0 + mi],      m0 + mi,      n0 + 16 + r, qb, kb, vb);
        kqv_scatter(c10[i] + bias[m0 + 16 + mi], m0 + 16 + mi, n0 + r,      qb, kb, vb);
        kqv_scatter(c11[i] + bias[m0 + 16 + mi], m0 + 16 + mi, n0 + 16 + r, qb, kb, vb);
    }
}

// -------- flash attention: 32 queries per wave, all lanes do softmax --------
__global__ void attention_kernel(const __bf16* __restrict__ qb,
                                 const __bf16* __restrict__ kb,
                                 const __bf16* __restrict__ vb,
                                 __bf16* __restrict__ aT)
{
    __shared__ float  S[32 * 33];                         // stride 33: no bank clash
    __shared__ __attribute__((aligned(16))) __bf16 P[32 * 32];
    __shared__ float  rmax[32], rsum[32], rscale[32];

    const int lane = threadIdx.x & 31;
    const int half = lane >> 4;
    const int r    = lane & 15;
    const int q0   = blockIdx.x * 32;
    const int head = blockIdx.y;

    const __bf16* qh = qb + head * 4096 * 32;
    const __bf16* kh = kb + head * 4096 * 32;
    const __bf16* vh = vb + head * 16 * 4096;

    v16bf qa0 = load_frag(qh + (q0 + r) * 32, half);
    v16bf qa1 = load_frag(qh + (q0 + 16 + r) * 32, half);

    rmax[lane] = -3.0e38f; rsum[lane] = 0.f;
    __syncthreads();

    v8f acc0 = {}, acc1 = {};
    for (int j0 = 0; j0 < 4096; j0 += 32) {
        v16bf kf0 = load_frag(kh + (j0 + r) * 32, half);
        v16bf kf1 = load_frag(kh + (j0 + 16 + r) * 32, half);
        v8f z = {};
        v8f s00 = wmma_bf16(qa0, kf0, z);
        v8f s01 = wmma_bf16(qa0, kf1, z);
        v8f s10 = wmma_bf16(qa1, kf0, z);
        v8f s11 = wmma_bf16(qa1, kf1, z);

        #pragma unroll
        for (int i = 0; i < 8; ++i) {
            int mi = i + 8 * half;
            S[mi * 33 + r]             = s00[i];
            S[mi * 33 + 16 + r]        = s01[i];
            S[(16 + mi) * 33 + r]      = s10[i];
            S[(16 + mi) * 33 + 16 + r] = s11[i];
        }
        __syncthreads();

        {   // online softmax: every lane owns one query row
            float mx = rmax[lane], tm = mx;
            for (int c = 0; c < 32; ++c) tm = fmaxf(tm, S[lane * 33 + c]);
            float sc  = __expf(mx - tm);
            float sum = rsum[lane] * sc;
            for (int c = 0; c < 32; ++c) {
                float p = __expf(S[lane * 33 + c] - tm);
                P[lane * 32 + c] = (__bf16)p;
                sum += p;
            }
            rmax[lane] = tm; rsum[lane] = sum; rscale[lane] = sc;
        }
        __syncthreads();

        #pragma unroll
        for (int i = 0; i < 8; ++i) {
            acc0[i] *= rscale[i + 8 * half];
            acc1[i] *= rscale[16 + i + 8 * half];
        }

        v16bf pa0 = load_frag(&P[r * 32], half);          // ds_load_b128 x2
        v16bf pa1 = load_frag(&P[(16 + r) * 32], half);
        v16bf vf  = load_frag(vh + r * 4096 + j0, half);  // col = dvh, K = key
        acc0 = wmma_bf16(pa0, vf, acc0);
        acc1 = wmma_bf16(pa1, vf, acc1);
        __syncthreads();
    }

    // store transposed bf16 at post-reshape (c,s): f = head*32768 + nq*8 + dvh
    if (r < 8) {
        #pragma unroll
        for (int i = 0; i < 8; ++i) {
            int mi = i + 8 * half;
            int f0 = ((head * 4096) + q0 + mi) * 8 + r;
            int f1 = ((head * 4096) + q0 + 16 + mi) * 8 + r;
            aT[(f0 & 4095) * 64 + (f0 >> 12)] = (__bf16)(acc0[i] / rsum[mi]);
            aT[(f1 & 4095) * 64 + (f1 >> 12)] = (__bf16)(acc1[i] / rsum[16 + mi]);
        }
    }
}

// ---- attn projection: [64x64] x [64x4096] -> out ch 192..255, 32x32/wave ---
__global__ void attn_proj_kernel(const __bf16* __restrict__ aT,
                                 const __bf16* __restrict__ wattn,
                                 const float* __restrict__ bias,
                                 float* __restrict__ out)
{
    const int lane = threadIdx.x & 31;
    const int half = lane >> 4;
    const int r    = lane & 15;
    const int m0   = blockIdx.x * 32;
    const int n0   = blockIdx.y * 32;

    const __bf16* a0 = wattn + (m0 + r) * 64;
    const __bf16* a1 = wattn + (m0 + 16 + r) * 64;
    const __bf16* b0 = aT    + (n0 + r) * 64;
    const __bf16* b1 = aT    + (n0 + 16 + r) * 64;

    v8f c00 = {}, c01 = {}, c10 = {}, c11 = {};
    #pragma unroll
    for (int k0 = 0; k0 < 64; k0 += 32) {
        v16bf af0 = load_frag(a0 + k0, half), af1 = load_frag(a1 + k0, half);
        v16bf bf0 = load_frag(b0 + k0, half), bf1 = load_frag(b1 + k0, half);
        c00 = wmma_bf16(af0, bf0, c00);
        c01 = wmma_bf16(af0, bf1, c01);
        c10 = wmma_bf16(af1, bf0, c10);
        c11 = wmma_bf16(af1, bf1, c11);
    }
    #pragma unroll
    for (int i = 0; i < 8; ++i) {
        int mi = i + 8 * half;
        out[(192 + m0 + mi) * 4096 + n0 + r]           = c00[i] + bias[m0 + mi];
        out[(192 + m0 + mi) * 4096 + n0 + 16 + r]      = c01[i] + bias[m0 + mi];
        out[(192 + m0 + 16 + mi) * 4096 + n0 + r]      = c10[i] + bias[m0 + 16 + mi];
        out[(192 + m0 + 16 + mi) * 4096 + n0 + 16 + r] = c11[i] + bias[m0 + 16 + mi];
    }
}

// --- 3x3x3 conv (pad 1) as implicit GEMM, out ch 0..191, 32x32 per wave -----
__global__ void conv3d_gemm_kernel(const __bf16* __restrict__ xT,
                                   const __bf16* __restrict__ wconv,
                                   const float* __restrict__ bias,
                                   float* __restrict__ out)
{
    const int lane = threadIdx.x & 31;
    const int half = lane >> 4;
    const int r    = lane & 15;
    const int m0   = blockIdx.x * 32;
    const int n0   = blockIdx.y * 32;

    const int na = n0 + r,      nb = n0 + 16 + r;
    const int ha = na >> 8, wa = (na >> 4) & 15, da = na & 15;
    const int hb = nb >> 8, wb_ = (nb >> 4) & 15, db = nb & 15;

    v8f c00 = {}, c01 = {}, c10 = {}, c11 = {};
    for (int t0 = 0; t0 < 3; ++t0)
    for (int t1 = 0; t1 < 3; ++t1)
    for (int t2 = 0; t2 < 3; ++t2) {
        int tap = (t0 * 3 + t1) * 3 + t2;

        int hsa = ha + t0 - 1, wsa = wa + t1 - 1, dsa = da + t2 - 1;
        int hsb = hb + t0 - 1, wsb = wb_ + t1 - 1, dsb = db + t2 - 1;
        bool va = ((unsigned)hsa < 16u) & ((unsigned)wsa < 16u) & ((unsigned)dsa < 16u);
        bool vb = ((unsigned)hsb < 16u) & ((unsigned)wsb < 16u) & ((unsigned)dsb < 16u);
        int nsa = (hsa << 8) + (wsa << 4) + dsa;
        int nsb = (hsb << 8) + (wsb << 4) + dsb;

        const __bf16* a0 = wconv + (tap * 192 + m0 + r) * 256;
        const __bf16* a1 = wconv + (tap * 192 + m0 + 16 + r) * 256;
        const __bf16* b0 = xT + nsa * 256;
        const __bf16* b1 = xT + nsb * 256;
        if (va) __builtin_prefetch(b0, 0, 3);            // global_prefetch_b8

        #pragma unroll
        for (int k0 = 0; k0 < 256; k0 += 32) {
            v16bf af0 = load_frag(a0 + k0, half);
            v16bf af1 = load_frag(a1 + k0, half);
            v16bf bf0 = {}, bf1 = {};
            if (va) bf0 = load_frag(b0 + k0, half);      // EXEC-masked; zero at border
            if (vb) bf1 = load_frag(b1 + k0, half);
            c00 = wmma_bf16(af0, bf0, c00);
            c01 = wmma_bf16(af0, bf1, c01);
            c10 = wmma_bf16(af1, bf0, c10);
            c11 = wmma_bf16(af1, bf1, c11);
        }
    }
    #pragma unroll
    for (int i = 0; i < 8; ++i) {
        int mi = i + 8 * half;
        out[(m0 + mi) * 4096 + na]      = c00[i] + bias[m0 + mi];
        out[(m0 + mi) * 4096 + nb]      = c01[i] + bias[m0 + mi];
        out[(m0 + 16 + mi) * 4096 + na] = c10[i] + bias[m0 + 16 + mi];
        out[(m0 + 16 + mi) * 4096 + nb] = c11[i] + bias[m0 + 16 + mi];
    }
}

extern "C" void kernel_launch(void* const* d_in, const int* in_sizes, int n_in,
                              void* d_out, int out_size, void* d_ws, size_t ws_size,
                              hipStream_t stream)
{
    const float* x      = (const float*)d_in[0];
    const float* kqv_w  = (const float*)d_in[1];
    const float* kqv_b  = (const float*)d_in[2];
    const float* attn_w = (const float*)d_in[3];
    const float* attn_b = (const float*)d_in[4];
    const float* conv_w = (const float*)d_in[5];
    const float* conv_b = (const float*)d_in[6];
    float* out = (float*)d_out;

    // bf16 workspace layout (all sub-buffers 16B aligned)
    __bf16* qb    = (__bf16*)d_ws;          // [8][4096][32]  1048576
    __bf16* kb    = qb    + 1048576;        // [8][4096][32]  1048576
    __bf16* vb    = kb    + 1048576;        // [8][16][4096]   524288  (vT)
    __bf16* xT    = vb    + 524288;         // [4096][256]    1048576
    __bf16* wkqv  = xT    + 1048576;        // [384][256]       98304
    __bf16* wconv = wkqv  + 98304;          // [27][192][256] 1327104
    __bf16* wattn = wconv + 1327104;        // [64][64]          4096
    __bf16* aT    = wattn + 4096;           // [4096][64]      262144

    const int nz = 1048576 * 2 + 524288;    // q+k+v (zero padding lanes)
    zero_bf16_kernel<<<(nz + 255) / 256, 256, 0, stream>>>(qb, nz);
    pack_xT_kernel<<<(1048576 + 255) / 256, 256, 0, stream>>>(x, xT);
    cvt_bf16_kernel<<<(98304 + 255) / 256, 256, 0, stream>>>(kqv_w, wkqv, 98304);
    pack_wconv_kernel<<<(1327104 + 255) / 256, 256, 0, stream>>>(conv_w, wconv);
    cvt_bf16_kernel<<<(4096 + 255) / 256, 256, 0, stream>>>(attn_w, wattn, 4096);

    kqv_gemm_kernel<<<dim3(12, 128), 32, 0, stream>>>(xT, wkqv, kqv_b, qb, kb, vb);
    attention_kernel<<<dim3(128, 8), 32, 0, stream>>>(qb, kb, vb, aT);
    attn_proj_kernel<<<dim3(2, 128), 32, 0, stream>>>(aT, wattn, attn_b, out);
    conv3d_gemm_kernel<<<dim3(6, 128), 32, 0, stream>>>(xT, wconv, conv_b, out);
}
